// SelfAttentionModel_35725537968569
// MI455X (gfx1250) — matrix operand order repaired and checked by reading, not compiled, not run on hardware
//
#include <hip/hip_runtime.h>
#include <cstddef>

#define B_   8
#define S_   2048
#define DIN  1024
#define DH   1024
#define DOUT 1024

typedef __bf16        v16bf __attribute__((ext_vector_type(16)));
typedef float         v8f   __attribute__((ext_vector_type(8)));
typedef unsigned int  u32x4 __attribute__((ext_vector_type(4)));

#define V8F_ZERO ((v8f){0.f,0.f,0.f,0.f,0.f,0.f,0.f,0.f})

union FragU { u32x4 u[2]; v16bf v; };

__device__ __forceinline__ unsigned short f2bf(float f) {
  unsigned u = __float_as_uint(f);
  u += 0x7FFFu + ((u >> 16) & 1u);        // round-to-nearest-even
  return (unsigned short)(u >> 16);
}
__device__ __forceinline__ float bf2f(unsigned short h) {
  return __uint_as_float(((unsigned)h) << 16);
}

__device__ __forceinline__ v8f wmma_bf16(v16bf a, v16bf b, v8f c) {
  // (neg_a, A, neg_b, B, c_mod, C, reuse_a, reuse_b)
  return __builtin_amdgcn_wmma_f32_16x16x32_bf16(false, a, false, b, (short)0, c, false, false);
}

// A fragment: 16x32 bf16 tile, rows row16..row16+15, K = kk..kk+31.
// lanes 0-15: row=lane, elems 0-7 at K+0..7, elems 8-15 at K+16..23
// lanes16-31: row=lane-16, elems at K+8..15 and K+24..31
__device__ __forceinline__ v16bf load_fragA(const unsigned short* base, int stride,
                                            int row16, int kk, int lane) {
  const unsigned short* p = base + (size_t)(row16 + (lane & 15)) * stride
                                 + kk + ((lane < 16) ? 0 : 8);
  FragU f;
  f.u[0] = *(const u32x4*)(p);
  f.u[1] = *(const u32x4*)(p + 16);
  return f.v;
}

// B fragment: 32x16 bf16 (K x N), source stored row-major as [N][K].
// lanes 0-15: N=lane, K = kk+0..7, kk+16..23; lanes 16-31: N=lane-16, K = kk+8..15, kk+24..31
__device__ __forceinline__ v16bf load_fragB(const unsigned short* base, int stride,
                                            int col16, int kk, int lane) {
  const unsigned short* p = base + (size_t)(col16 + (lane & 15)) * stride
                                 + kk + ((lane < 16) ? 0 : 8);
  FragU f;
  f.u[0] = *(const u32x4*)(p);
  f.u[1] = *(const u32x4*)(p + 16);
  return f.v;
}

// ---------------- elementwise helpers ----------------

__global__ void cvt_bf16_4(const float* __restrict__ src, unsigned short* __restrict__ dst, int n4) {
  int i = blockIdx.x * blockDim.x + threadIdx.x;
  if (i >= n4) return;
  float4 f = ((const float4*)src)[i];
  ushort4 h;
  h.x = f2bf(f.x); h.y = f2bf(f.y); h.z = f2bf(f.z); h.w = f2bf(f.w);
  ((ushort4*)dst)[i] = h;
}

__global__ void zero_f32(float* __restrict__ p, int n) {
  int i = blockIdx.x * blockDim.x + threadIdx.x;
  if (i < n) p[i] = 0.0f;
}

// ---------------- QKV projection GEMM ----------------
// grid: (ROWS/128, DH/128, 3), block: 128 (4 waves), each wave owns a 64x64 tile.
// mode 0: Q = (x Wq^T + bq) * 1/sqrt(DH), stored [b*s][h] bf16
// mode 1: K = x Wk^T + bk,                stored [b*s][h] bf16
// mode 2: V^T = (x Wv^T + bv)^T,          stored [b][h][s] bf16
__global__ __launch_bounds__(128) void qkv_gemm(
    const unsigned short* __restrict__ xb,
    const unsigned short* __restrict__ wqb,
    const unsigned short* __restrict__ wkb,
    const unsigned short* __restrict__ wvb,
    const float* __restrict__ bq, const float* __restrict__ bk, const float* __restrict__ bv,
    unsigned short* __restrict__ Q, unsigned short* __restrict__ Ko, unsigned short* __restrict__ Vt)
{
  const int lane = threadIdx.x & 31;
  const int wid  = threadIdx.x >> 5;
  const int mode = blockIdx.z;
  const unsigned short* W = (mode == 0) ? wqb : (mode == 1) ? wkb : wvb;
  const float* bias = (mode == 0) ? bq : (mode == 1) ? bk : bv;

  const int r0 = blockIdx.x * 128 + (wid >> 1) * 64;
  const int c0 = blockIdx.y * 128 + (wid & 1) * 64;

  v8f acc[4][4];
  #pragma unroll
  for (int i = 0; i < 4; ++i)
    #pragma unroll
    for (int j = 0; j < 4; ++j) acc[i][j] = V8F_ZERO;

  for (int kk = 0; kk < DIN; kk += 32) {
    v16bf a[4], bb[4];
    #pragma unroll
    for (int i = 0; i < 4; ++i) a[i]  = load_fragA(xb, DIN, r0 + 16*i, kk, lane);
    #pragma unroll
    for (int j = 0; j < 4; ++j) bb[j] = load_fragB(W,  DIN, c0 + 16*j, kk, lane);
    #pragma unroll
    for (int i = 0; i < 4; ++i)
      #pragma unroll
      for (int j = 0; j < 4; ++j)
        acc[i][j] = wmma_bf16(a[i], bb[j], acc[i][j]);
  }

  const float scale = (mode == 0) ? 0.03125f : 1.0f;  // 1/sqrt(1024)
  #pragma unroll
  for (int j = 0; j < 4; ++j) {
    const int col = c0 + 16*j + (lane & 15);
    const float bv_ = bias[col];
    #pragma unroll
    for (int i = 0; i < 4; ++i) {
      if (mode < 2) {
        unsigned short* Out = (mode == 0) ? Q : Ko;
        #pragma unroll
        for (int r = 0; r < 8; ++r) {
          int row = r0 + 16*i + ((lane < 16) ? r : 8 + r);
          Out[(size_t)row * DH + col] = f2bf((acc[i][j][r] + bv_) * scale);
        }
      } else {
        // V^T: components r map to consecutive s positions -> one b128 store
        int rowt = r0 + 16*i;
        int bidx = rowt >> 11;              // / 2048
        int s0   = (rowt & 2047) + ((lane < 16) ? 0 : 8);
        u32x4 pk;
        #pragma unroll
        for (int r = 0; r < 4; ++r) {
          unsigned lo = f2bf(acc[i][j][2*r]     + bv_);
          unsigned hi = f2bf(acc[i][j][2*r + 1] + bv_);
          pk[r] = lo | (hi << 16);
        }
        *(u32x4*)(Vt + ((size_t)(bidx * DH + col)) * S_ + s0) = pk;
      }
    }
  }
}

// ---------------- fused attention ----------------
// grid: B*S/16 single-wave workgroups. Each wave: one (b, 16-query) tile.
__global__ __launch_bounds__(32) void attn_kernel(
    const unsigned short* __restrict__ Qb,
    const unsigned short* __restrict__ Kb,
    const unsigned short* __restrict__ Vt,
    const long long* __restrict__ lens,
    float* __restrict__ Osum)
{
  const int PSTR = 2056;                       // 2048 + 8 pad: conflict-free b128 frag reads
  __shared__ unsigned short Pl[16 * 2056];     // 64.25 KB score/prob buffer
  __shared__ float Ml[16], Ll[16];

  const int lane = threadIdx.x;
  const int qt = blockIdx.x;
  const int b  = qt >> 7;
  const int q0 = (qt & 127) << 4;
  const int gq = b * S_ + q0;
  const int mlen = (int)lens[b];
  const int col_lane = lane & 15;
  const int rbase = (lane < 16) ? 0 : 8;

  // Pin Q tile (16 x 1024) as 32 A-fragments (256 VGPRs)
  v16bf qf[32];
  #pragma unroll
  for (int t = 0; t < 32; ++t) qf[t] = load_fragA(Qb, DH, gq, t * 32, lane);

  float m_p[8], l_p[8];
  #pragma unroll
  for (int r = 0; r < 8; ++r) { m_p[r] = -3.0e38f; l_p[r] = 0.0f; }

  // Phase A: scores -> LDS (bf16), per-lane online (m,l) partials.
  // Two key tiles in flight -> two independent WMMA accumulation chains
  // (occupancy is 1 wave/SIMD here, so ILP must come from within the wave).
  for (int kb = 0; kb < S_ / 16; kb += 2) {
    const int gk0 = b * S_ + kb * 16;
    const int gk1 = gk0 + 16;
    v8f acc0 = V8F_ZERO, acc1 = V8F_ZERO;
    #pragma unroll
    for (int t = 0; t < 32; ++t) {
      v16bf kf0 = load_fragB(Kb, DH, gk0, t * 32, lane);
      v16bf kf1 = load_fragB(Kb, DH, gk1, t * 32, lane);
      acc0 = wmma_bf16(qf[t], kf0, acc0);
      acc1 = wmma_bf16(qf[t], kf1, acc1);
    }
    #pragma unroll
    for (int half = 0; half < 2; ++half) {
      const v8f acc = half ? acc1 : acc0;
      const int key = (kb + half) * 16 + col_lane;
      const bool valid = key < mlen;
      #pragma unroll
      for (int r = 0; r < 8; ++r) {
        float s = valid ? acc[r] : -1.0e9f;
        Pl[(rbase + r) * PSTR + (kb + half) * 16 + col_lane] = f2bf(s);
        if (valid) {
          float mo = m_p[r];
          float mn = fmaxf(mo, s);
          l_p[r] = l_p[r] * __expf(mo - mn) + __expf(s - mn);
          m_p[r] = mn;
        }
      }
    }
  }

  // Exact (m,l) per row: butterfly within each 16-lane half
  #pragma unroll
  for (int r = 0; r < 8; ++r) {
    float m = m_p[r], l = l_p[r];
    #pragma unroll
    for (int off = 1; off < 16; off <<= 1) {
      float mo = __shfl_xor(m, off, 32);
      float lo = __shfl_xor(l, off, 32);
      float mn = fmaxf(m, mo);
      l = l * __expf(m - mn) + lo * __expf(mo - mn);
      m = mn;
    }
    m_p[r] = m; l_p[r] = l;
  }
  if (lane == 0 || lane == 16) {
    #pragma unroll
    for (int r = 0; r < 8; ++r) { Ml[rbase + r] = m_p[r]; Ll[rbase + r] = l_p[r]; }
  }
  __syncthreads();
  const float mrow = Ml[col_lane];
  const float invl = 1.0f / Ll[col_lane];

  // Phase A.5: P = exp(s - m) / l, in place. Each lane's A-frag elems are one row.
  {
    unsigned short* prow = &Pl[col_lane * PSTR];
    for (int kb = 0; kb < S_ / 32; ++kb) {
      unsigned short* p = prow + kb * 32 + ((lane < 16) ? 0 : 8);
      FragU f;
      f.u[0] = *(u32x4*)(p);
      f.u[1] = *(u32x4*)(p + 16);
      unsigned short* e = (unsigned short*)&f;
      #pragma unroll
      for (int i = 0; i < 16; ++i)
        e[i] = f2bf(__expf(bf2f(e[i]) - mrow) * invl);
      *(u32x4*)(p)      = f.u[0];
      *(u32x4*)(p + 16) = f.u[1];
    }
  }
  __syncthreads();

  // Phase B: O = P * V, P fragment reused across 8 h-tiles (8 independent
  // WMMA chains); column-sum over the 16 queries -> atomic add into Osum.
  const unsigned short* Vtb = Vt + (size_t)b * DH * S_;
  float* Ob = Osum + b * DH;
  for (int hc = 0; hc < 8; ++hc) {
    v8f o[8];
    #pragma unroll
    for (int j = 0; j < 8; ++j) o[j] = V8F_ZERO;
    for (int kb = 0; kb < S_ / 32; ++kb) {
      const unsigned short* p = &Pl[col_lane * PSTR] + kb * 32 + ((lane < 16) ? 0 : 8);
      FragU pf;
      pf.u[0] = *(const u32x4*)(p);
      pf.u[1] = *(const u32x4*)(p + 16);
      #pragma unroll
      for (int j = 0; j < 8; ++j) {
        v16bf vf = load_fragB(Vtb, S_, hc * 128 + j * 16, kb * 32, lane);
        o[j] = wmma_bf16(pf.v, vf, o[j]);
      }
    }
    #pragma unroll
    for (int j = 0; j < 8; ++j) {
      float cs = 0.0f;
      #pragma unroll
      for (int r = 0; r < 8; ++r) cs += o[j][r];
      cs += __shfl_xor(cs, 16, 32);
      if (lane < 16) atomicAdd(&Ob[hc * 128 + j * 16 + lane], cs);
    }
  }
}

// ---------------- final projection + mean ----------------
__global__ void out_proj_kernel(const float* __restrict__ Osum,
                                const float* __restrict__ Wo,
                                const float* __restrict__ bo,
                                float* __restrict__ out)
{
  int t = blockIdx.x * blockDim.x + threadIdx.x;
  if (t >= B_ * DOUT) return;
  int b = t / DOUT, o = t - b * DOUT;
  const float4* orow = (const float4*)(Osum + b * DH);
  const float4* wrow = (const float4*)(Wo + (size_t)o * DH);
  float s = 0.0f;
  #pragma unroll 2
  for (int h = 0; h < DH / 4; ++h) {
    float4 a = orow[h], w = wrow[h];
    s += a.x * w.x + a.y * w.y + a.z * w.z + a.w * w.w;
  }
  out[t] = bo[o] + s * (1.0f / (float)S_);
}

extern "C" void kernel_launch(void* const* d_in, const int* in_sizes, int n_in,
                              void* d_out, int out_size, void* d_ws, size_t ws_size,
                              hipStream_t stream) {
  const float*     x    = (const float*)d_in[0];
  const long long* lens = (const long long*)d_in[1];
  const float*     Wq   = (const float*)d_in[2];
  const float*     bq   = (const float*)d_in[3];
  const float*     Wk   = (const float*)d_in[4];
  const float*     bk   = (const float*)d_in[5];
  const float*     Wv   = (const float*)d_in[6];
  const float*     bv   = (const float*)d_in[7];
  const float*     Wo   = (const float*)d_in[8];
  const float*     bo   = (const float*)d_in[9];
  float* out = (float*)d_out;

  char* ws = (char*)d_ws;
  size_t off = 0;
  const size_t XSZ = (size_t)B_ * S_ * DIN * 2;   // 33.5 MB bf16
  const size_t WSZ = (size_t)DH * DIN * 2;        // 2 MB bf16
  unsigned short* xb  = (unsigned short*)(ws + off); off += XSZ;
  unsigned short* Qb  = (unsigned short*)(ws + off); off += XSZ;
  unsigned short* Kb  = (unsigned short*)(ws + off); off += XSZ;
  unsigned short* Vt  = (unsigned short*)(ws + off); off += XSZ;
  unsigned short* wqb = (unsigned short*)(ws + off); off += WSZ;
  unsigned short* wkb = (unsigned short*)(ws + off); off += WSZ;
  unsigned short* wvb = (unsigned short*)(ws + off); off += WSZ;
  float* Osum = (float*)(ws + off); off += (size_t)B_ * DH * 4;

  // 1) fp32 -> bf16 conversions
  {
    int n4 = (B_ * S_ * DIN) / 4;
    cvt_bf16_4<<<n4 / 256, 256, 0, stream>>>(x, xb, n4);
    int w4 = (DH * DIN) / 4;
    cvt_bf16_4<<<w4 / 256, 256, 0, stream>>>(Wq, wqb, w4);
    cvt_bf16_4<<<w4 / 256, 256, 0, stream>>>(Wk, wkb, w4);
    cvt_bf16_4<<<w4 / 256, 256, 0, stream>>>(Wv, wvb, w4);
  }
  // 2) zero attention-output accumulator (ws is poisoned by harness)
  zero_f32<<<(B_ * DH) / 256, 256, 0, stream>>>(Osum, B_ * DH);

  // 3) QKV projections (z: 0=Q,1=K,2=V^T)
  {
    dim3 grid((B_ * S_) / 128, DH / 128, 3);
    qkv_gemm<<<grid, 128, 0, stream>>>(xb, wqb, wkb, wvb, bq, bk, bv, Qb, Kb, Vt);
  }
  // 4) fused attention (single-wave workgroups, one per 16-query tile)
  attn_kernel<<<(B_ * S_) / 16, 32, 0, stream>>>(Qb, Kb, Vt, lens, Osum);

  // 5) output projection folded with mean over sequence
  out_proj_kernel<<<(B_ * DOUT + 255) / 256, 256, 0, stream>>>(Osum, Wo, bo, out);
}
